// TAttention_34935263985839
// MI455X (gfx1250) — matrix operand order repaired and checked by reading, not compiled, not run on hardware
//
#include <hip/hip_runtime.h>
#include <hip/hip_bf16.h>

// ---------------------------------------------------------------------------
// Types for WMMA operands (CDNA5 / gfx1250, wave32)
// ---------------------------------------------------------------------------
typedef __attribute__((ext_vector_type(16))) __bf16          v16bf;
typedef __attribute__((ext_vector_type(8)))  float           v8f;
typedef __attribute__((ext_vector_type(8)))  unsigned short  v8us;
typedef __attribute__((ext_vector_type(4)))  unsigned int    v4u;
typedef __attribute__((ext_vector_type(8)))  unsigned int    v8u;

union Frag {
    v16bf bf;
    v8us  h[2];
};

__device__ __forceinline__ unsigned short f2bf(float f) {
    unsigned int u = __float_as_uint(f);
    unsigned int r = u + 0x7FFFu + ((u >> 16) & 1u);   // round-to-nearest-even
    return (unsigned short)(r >> 16);
}

__device__ __forceinline__ v8f vzero8() {
    v8f z;
#pragma unroll
    for (int i = 0; i < 8; ++i) z[i] = 0.0f;
    return z;
}

// ---------------------------------------------------------------------------
// Tensor Data Mover: async 2-D tile DMA global -> LDS (CDNA5 TDM, D# per
// CDNA5 ISA). Issued once per wave; EXEC ignored;
// completion tracked with TENSORcnt. pad_interval/pad_amount insert LDS
// padding so the destination tile has a bank-conflict-free stride.
//   pad codes here: interval=4 (32 DWORDs = 128B row), amount=3 (4 DWORDs).
// ---------------------------------------------------------------------------
__device__ __forceinline__ void tdm_load_2d(
    unsigned lds_off,          // LDS byte offset of tile start
    const void* gaddr,         // global address of tile start
    unsigned tensor_d0,        // tensor length dim0 (elements)
    unsigned tensor_d1,        // tensor length dim1 (rows)
    unsigned tile_d0,          // tile width  (elements)
    unsigned tile_d1,          // tile height (rows)
    unsigned stride_d0,        // row stride  (elements)
    unsigned pad_interval,     // code: DWORDs before pad = 2<<code
    unsigned pad_amount)       // code: pad DWORDs = code+1
{
    const unsigned long long ga = (unsigned long long)gaddr;
    v4u g0;
    g0[0] = 1u;                                   // count=1 (valid), user mode
    g0[1] = lds_off;                              // lds_addr
    g0[2] = (unsigned)(ga & 0xFFFFFFFFu);         // global_addr[31:0]
    g0[3] = (unsigned)((ga >> 32) & 0x1FFFFFFu)   // global_addr[56:32]
          | (2u << 30);                           // type=2 ("image")
    v8u g1;
    g1[0] = (1u << 16)                            // data_size = 2 bytes
          | (1u << 20)                            // pad_enable
          | (pad_interval << 22)
          | (pad_amount << 25);
    g1[1] = (tensor_d0 & 0xFFFFu) << 16;          // tensor_dim0[15:0]
    g1[2] = (tensor_d0 >> 16) | ((tensor_d1 & 0xFFFFu) << 16);
    g1[3] = (tensor_d1 >> 16) | (tile_d0 << 16);  // tensor_dim1 hi | tile_dim0
    g1[4] = tile_d1;                              // tile_dim1 (tile_dim2 = 0)
    g1[5] = stride_d0;                            // tensor_dim0_stride[31:0]
    g1[6] = 0u;                                   // stride hi | dim1 stride lo
    g1[7] = 0u;
    v4u gz;
    gz[0] = 0u; gz[1] = 0u; gz[2] = 0u; gz[3] = 0u;
    asm volatile("tensor_load_to_lds %0, %1, %2, %3"
                 :: "s"(g0), "s"(g1), "s"(gz), "s"(gz)
                 : "memory");
}

__device__ __forceinline__ unsigned lds_offset_of(const void* p) {
    // Generic LDS address = {SHARED_BASE aperture, 32-bit LDS byte offset}.
    return (unsigned)(unsigned long long)p;
}

// ---------------------------------------------------------------------------
// fp32 -> bf16 elementwise cast (for weights)
// ---------------------------------------------------------------------------
__global__ __launch_bounds__(256) void cast_bf16_kernel(
    const float* __restrict__ in, unsigned short* __restrict__ out, int n) {
    int i = blockIdx.x * 256 + threadIdx.x;
    if (i < n) out[i] = f2bf(in[i]);
}

// ---------------------------------------------------------------------------
// LayerNorm over D=1024 (one block per row), optional residual add.
// ---------------------------------------------------------------------------
template <bool HAS_RES>
__global__ __launch_bounds__(256) void ln_kernel(
    const float* __restrict__ x, const float* __restrict__ res,
    const float* __restrict__ gamma, const float* __restrict__ beta,
    float* __restrict__ outF, unsigned short* __restrict__ outB) {
    const int D = 1024;
    const size_t base = (size_t)blockIdx.x * D;
    float v[4];
    float s = 0.f, s2 = 0.f;
#pragma unroll
    for (int i = 0; i < 4; ++i) {
        int idx = threadIdx.x + i * 256;
        float t = x[base + idx];
        if (HAS_RES) t += res[base + idx];
        v[i] = t;
        s += t;
        s2 += t * t;
    }
#pragma unroll
    for (int off = 16; off > 0; off >>= 1) {
        s  += __shfl_down(s,  off, 32);
        s2 += __shfl_down(s2, off, 32);
    }
    __shared__ float rs[8], rs2[8];
    int lane = threadIdx.x & 31, wv = threadIdx.x >> 5;
    if (lane == 0) { rs[wv] = s; rs2[wv] = s2; }
    __syncthreads();
    float ts = 0.f, ts2 = 0.f;
#pragma unroll
    for (int i = 0; i < 8; ++i) { ts += rs[i]; ts2 += rs2[i]; }
    const float mu   = ts * (1.0f / 1024.0f);
    const float var  = ts2 * (1.0f / 1024.0f) - mu * mu;
    const float rstd = rsqrtf(var + 1e-5f);
#pragma unroll
    for (int i = 0; i < 4; ++i) {
        int idx = threadIdx.x + i * 256;
        float o = (v[i] - mu) * rstd * gamma[idx] + beta[idx];
        outF[base + idx] = o;
        outB[base + idx] = f2bf(o);
    }
}

// ---------------------------------------------------------------------------
// bf16 WMMA GEMM:  C[M,N] = A[M,K] @ W[N,K]^T  (torch Linear convention)
// Block tile 128x128, K-step 64, 8 waves (wave tile 32x64 = 2x4 accum tiles).
// A/W tiles DMA'd into double-buffered LDS by the Tensor Data Mover; the
// next tile's DMA overlaps the current tile's 16 WMMAs per wave.
// ---------------------------------------------------------------------------
#define TILE_M 128
#define TILE_N 128
#define TILE_K 64
#define BKP    72   // padded LDS K-stride (ushorts): 144B (TDM pad 16B/row)

template <bool OUT_BF16, bool HAS_BIAS, bool RELU, bool HAS_RES>
__global__ __launch_bounds__(256) void gemm_bf16_wmma(
    const unsigned short* __restrict__ A,    // [M,K] bf16
    const unsigned short* __restrict__ Bw,   // [N,K] bf16 (weight rows)
    const float* __restrict__ bias,          // [N] or null
    const float* __restrict__ res,           // [M,N] or null
    float* __restrict__ Cf,                  // [M,N] fp32 out (if !OUT_BF16)
    unsigned short* __restrict__ Cb,         // [M,N] bf16 out (if OUT_BF16)
    int M, int N, int K) {
    __shared__ __align__(16) unsigned short As[2][TILE_M * BKP];
    __shared__ __align__(16) unsigned short Bs[2][TILE_N * BKP];

    const int tid  = threadIdx.x;
    const int lane = tid & 31;
    const int wv   = tid >> 5;            // wave 0..7
    const int m0b  = blockIdx.y * TILE_M;
    const int n0b  = blockIdx.x * TILE_N;
    const int wm   = (wv & 3) * 32;       // wave M offset (two 16-row frags)
    const int wn   = (wv >> 2) * 64;      // wave N offset (four 16-col frags)
    const int lr   = lane & 15;
    const int hi   = lane >> 4;           // lane half

    v8f acc[2][4];
#pragma unroll
    for (int i = 0; i < 2; ++i)
#pragma unroll
        for (int j = 0; j < 4; ++j) acc[i][j] = vzero8();

    // TDM issue for one K-stage (both tiles), wave 0 only.
    auto issue = [&](int buf, int k0) {
        if (wv == 0) {
            tdm_load_2d(lds_offset_of(&As[buf][0]),
                        A + (size_t)m0b * K + k0,
                        (unsigned)K, (unsigned)M, TILE_K, TILE_M,
                        (unsigned)K, 4u, 3u);
            tdm_load_2d(lds_offset_of(&Bs[buf][0]),
                        Bw + (size_t)n0b * K + k0,
                        (unsigned)K, (unsigned)N, TILE_K, TILE_N,
                        (unsigned)K, 4u, 3u);
        }
    };

    const int nIter = K / TILE_K;
    issue(0, 0);
    for (int it = 0; it < nIter; ++it) {
        const int buf = it & 1;
        if (it + 1 < nIter) {
            issue(buf ^ 1, (it + 1) * TILE_K);
            // 2 newest DMAs may remain outstanding; current stage complete.
            __builtin_amdgcn_s_wait_tensorcnt(2);
        } else {
            __builtin_amdgcn_s_wait_tensorcnt(0);
        }
        __syncthreads();

#pragma unroll
        for (int ks = 0; ks < TILE_K; ks += 32) {
            // A-frag (16-bit 16x32): lanes 0-15 K={0..7,16..23},
            // lanes 16-31 K={8..15,24..31}; row = lane&15.
            Frag a[2];
#pragma unroll
            for (int i = 0; i < 2; ++i) {
                int row = wm + i * 16 + lr;
                int kb  = ks + (hi ? 8 : 0);
                a[i].h[0] = *(const v8us*)&As[buf][row * BKP + kb];
                a[i].h[1] = *(const v8us*)&As[buf][row * BKP + kb + 16];
            }
            // B-frag (16-bit 32x16): lanes 0-15 K={0..7,16..23}, lanes 16-31 K={8..15,24..31}.
            Frag b[4];
#pragma unroll
            for (int j = 0; j < 4; ++j) {
                int col = wn + j * 16 + lr;
                int kb  = ks + (hi ? 8 : 0);
                b[j].h[0] = *(const v8us*)&Bs[buf][col * BKP + kb];
                b[j].h[1] = *(const v8us*)&Bs[buf][col * BKP + kb + 16];
            }
#pragma unroll
            for (int i = 0; i < 2; ++i)
#pragma unroll
                for (int j = 0; j < 4; ++j)
                    acc[i][j] = __builtin_amdgcn_wmma_f32_16x16x32_bf16(
                        false, a[i].bf, false, b[j].bf, (short)0, acc[i][j],
                        false, false);
        }
        __syncthreads();
    }

    // Epilogue: C layout VGPR r -> (M = r + 8*hi, N = lane&15).
#pragma unroll
    for (int i = 0; i < 2; ++i) {
#pragma unroll
        for (int j = 0; j < 4; ++j) {
            const int gcol = n0b + wn + j * 16 + lr;
            const float bs = HAS_BIAS ? bias[gcol] : 0.0f;
#pragma unroll
            for (int r = 0; r < 8; ++r) {
                const int grow = m0b + wm + i * 16 + r + 8 * hi;
                float v = acc[i][j][r] + bs;
                if (HAS_RES) v += res[(size_t)grow * N + gcol];
                if (RELU)    v = fmaxf(v, 0.0f);
                if (OUT_BF16) Cb[(size_t)grow * N + gcol] = f2bf(v);
                else          Cf[(size_t)grow * N + gcol] = v;
            }
        }
    }
}

// ---------------------------------------------------------------------------
// Flash attention (no 1/sqrt(dh) scaling, faithful to reference).
// Grid: (S/64, B*H). Block: 128 threads = 4 waves, each wave owns 16 q rows.
// K tile staged by TDM (row-major, padded); V staged transposed manually.
// Per 32-key chunk: 4 WMMAs for scores, online softmax, 4 WMMAs for P@V.
// ---------------------------------------------------------------------------
#define DH 64
#define KC 32

__global__ __launch_bounds__(128) void flash_attn_kernel(
    const unsigned short* __restrict__ Qb,   // [B*S, D] bf16
    const unsigned short* __restrict__ Kb,   // [B*S, D] bf16
    const unsigned short* __restrict__ Vb,   // [B*S, D] bf16
    float* __restrict__ Att,                 // [B*S, D] fp32
    int Hn, int S, int D) {
    const int bh   = blockIdx.y;
    const int b    = bh / Hn;
    const int h    = bh % Hn;
    const int qt   = blockIdx.x;
    const int tid  = threadIdx.x;
    const int lane = tid & 31;
    const int wv   = tid >> 5;      // 0..3
    const int lr   = lane & 15;
    const int hi   = lane >> 4;

    __shared__ __align__(16) unsigned short Klds[KC][72];     // [key][d], pad
    __shared__ __align__(16) unsigned short Vt[DH][40];       // [d][key], pad
    __shared__ __align__(16) unsigned short Plds[4][16][40];  // per-wave P

    const size_t qbase = ((size_t)(b * S + qt * 64 + wv * 16)) * D + h * DH;

    // Q fragments: 16x64 bf16 as two A-frags (K chunks d=0..31, 32..63).
    Frag qa[2];
#pragma unroll
    for (int c = 0; c < 2; ++c) {
        const int kb = hi ? 8 : 0;
        const unsigned short* p = Qb + qbase + (size_t)lr * D + c * 32 + kb;
        qa[c].h[0] = *(const v8us*)p;
        qa[c].h[1] = *(const v8us*)(p + 16);
    }

    float mrow[8], lrow[8];
    v8f o[4];
#pragma unroll
    for (int r = 0; r < 8; ++r) { mrow[r] = -3.0e38f; lrow[r] = 0.0f; }
#pragma unroll
    for (int n = 0; n < 4; ++n) o[n] = vzero8();

    for (int j0 = 0; j0 < S; j0 += KC) {
        // K chunk [32 keys x 64 d] via TDM (pad 16B per 128B row -> stride 72).
        if (wv == 0) {
            tdm_load_2d(lds_offset_of(&Klds[0][0]),
                        Kb + ((size_t)(b * S + j0)) * D + h * DH,
                        (unsigned)D, 0x40000000u, DH, KC,
                        (unsigned)D, 4u, 3u);
        }
        // V chunk transposed [64 d x 32 keys], manual scatter.
#pragma unroll
        for (int i = 0; i < 2; ++i) {
            int c   = tid + i * 128;      // 0..255 chunks of 8 ushorts
            int key = c >> 3;
            int dc  = (c & 7) * 8;
            v8us vvv = *(const v8us*)
                &Vb[((size_t)(b * S + j0 + key)) * D + h * DH + dc];
#pragma unroll
            for (int e = 0; e < 8; ++e) Vt[dc + e][key] = vvv[e];
        }
        __builtin_amdgcn_s_wait_tensorcnt(0);
        __syncthreads();

        // Scores: two 16(q) x 16(key) tiles, K-dim = dh = 64 (2 WMMAs each).
        v8f sc[2];
#pragma unroll
        for (int t = 0; t < 2; ++t) {
            v8f s = vzero8();
#pragma unroll
            for (int c = 0; c < 2; ++c) {
                Frag kf;
                const int kk = hi ? 8 : 0;
                kf.h[0] = *(const v8us*)&Klds[t * 16 + lr][c * 32 + kk];
                kf.h[1] = *(const v8us*)&Klds[t * 16 + lr][c * 32 + kk + 16];
                s = __builtin_amdgcn_wmma_f32_16x16x32_bf16(
                    false, qa[c].bf, false, kf.bf, (short)0, s, false, false);
            }
            sc[t] = s;
        }

        // Online softmax; rows live across the 16 lanes of each half.
        float p0[8], p1[8], scl[8];
#pragma unroll
        for (int r = 0; r < 8; ++r) {
            float mx = fmaxf(sc[0][r], sc[1][r]);
#pragma unroll
            for (int off = 1; off < 16; off <<= 1)
                mx = fmaxf(mx, __shfl_xor(mx, off, 32));
            const float mn = fmaxf(mrow[r], mx);
            const float e0 = __expf(sc[0][r] - mn);
            const float e1 = __expf(sc[1][r] - mn);
            float rsum = e0 + e1;
#pragma unroll
            for (int off = 1; off < 16; off <<= 1)
                rsum += __shfl_xor(rsum, off, 32);
            const float sc_ = __expf(mrow[r] - mn);
            lrow[r] = lrow[r] * sc_ + rsum;
            mrow[r] = mn;
            scl[r]  = sc_;
            p0[r] = e0;
            p1[r] = e1;
        }
#pragma unroll
        for (int n = 0; n < 4; ++n)
#pragma unroll
            for (int r = 0; r < 8; ++r) o[n][r] *= scl[r];

        // P (C layout) -> per-wave LDS -> A layout fragment.
#pragma unroll
        for (int r = 0; r < 8; ++r) {
            const int row = r + 8 * hi;
            Plds[wv][row][lr]      = f2bf(p0[r]);
            Plds[wv][row][16 + lr] = f2bf(p1[r]);
        }
        Frag pa;
        {
            const int kb = hi ? 8 : 0;
            pa.h[0] = *(const v8us*)&Plds[wv][lr][kb];
            pa.h[1] = *(const v8us*)&Plds[wv][lr][kb + 16];
        }

        // o += P(16x32) @ V(32x64): four 16-col d tiles.
#pragma unroll
        for (int n = 0; n < 4; ++n) {
            Frag vf;
            const int kb = hi ? 8 : 0;
            vf.h[0] = *(const v8us*)&Vt[n * 16 + lr][kb];
            vf.h[1] = *(const v8us*)&Vt[n * 16 + lr][kb + 16];
            o[n] = __builtin_amdgcn_wmma_f32_16x16x32_bf16(
                false, pa.bf, false, vf.bf, (short)0, o[n], false, false);
        }
        __syncthreads();
    }

    // Normalize and write fp32 attention output (head-interleaved layout).
#pragma unroll
    for (int n = 0; n < 4; ++n) {
#pragma unroll
        for (int r = 0; r < 8; ++r) {
            const int row = r + 8 * hi;
            Att[qbase + (size_t)row * D + n * 16 + lr] = o[n][r] / lrow[r];
        }
    }
}

// ---------------------------------------------------------------------------
// Host-side launch
// ---------------------------------------------------------------------------
extern "C" void kernel_launch(void* const* d_in, const int* in_sizes, int n_in,
                              void* d_out, int out_size, void* d_ws,
                              size_t ws_size, hipStream_t stream) {
    const float* x    = (const float*)d_in[0];
    const float* Wq   = (const float*)d_in[1];
    const float* Wk   = (const float*)d_in[2];
    const float* Wv   = (const float*)d_in[3];
    const float* W1   = (const float*)d_in[4];
    const float* b1   = (const float*)d_in[5];
    const float* W2   = (const float*)d_in[6];
    const float* b2   = (const float*)d_in[7];
    const float* ln1s = (const float*)d_in[8];
    const float* ln1b = (const float*)d_in[9];
    const float* ln2s = (const float*)d_in[10];
    const float* ln2b = (const float*)d_in[11];

    const int B = 4, S = 2048, D = 1024, H = 16;
    const int M = B * S;  // 8192

    char* ws = (char*)d_ws;
    size_t off = 0;
    auto alloc = [&](size_t bytes) -> void* {
        void* p = ws + off;
        off = (off + bytes + 255) & ~(size_t)255;
        return p;
    };

    float* xn  = (float*)alloc((size_t)M * D * 4);
    float* att = (float*)alloc((size_t)M * D * 4);
    float* xt  = (float*)alloc((size_t)M * D * 4);
    unsigned short* xnb = (unsigned short*)alloc((size_t)M * D * 2);
    unsigned short* xtb = (unsigned short*)alloc((size_t)M * D * 2);
    unsigned short* qb  = (unsigned short*)alloc((size_t)M * D * 2);
    unsigned short* kb  = (unsigned short*)alloc((size_t)M * D * 2);
    unsigned short* vb  = (unsigned short*)alloc((size_t)M * D * 2);
    unsigned short* hb  = (unsigned short*)alloc((size_t)M * D * 2);
    unsigned short* wqb = (unsigned short*)alloc((size_t)D * D * 2);
    unsigned short* wkb = (unsigned short*)alloc((size_t)D * D * 2);
    unsigned short* wvb = (unsigned short*)alloc((size_t)D * D * 2);
    unsigned short* w1b = (unsigned short*)alloc((size_t)D * D * 2);
    unsigned short* w2b = (unsigned short*)alloc((size_t)D * D * 2);

    const int nW = D * D;
    const int gc = (nW + 255) / 256;
    cast_bf16_kernel<<<gc, 256, 0, stream>>>(Wq, wqb, nW);
    cast_bf16_kernel<<<gc, 256, 0, stream>>>(Wk, wkb, nW);
    cast_bf16_kernel<<<gc, 256, 0, stream>>>(Wv, wvb, nW);
    cast_bf16_kernel<<<gc, 256, 0, stream>>>(W1, w1b, nW);
    cast_bf16_kernel<<<gc, 256, 0, stream>>>(W2, w2b, nW);

    // LN1
    ln_kernel<false><<<M, 256, 0, stream>>>(x, nullptr, ln1s, ln1b, xn, xnb);

    // q,k,v projections (bf16 out)
    dim3 gg(D / TILE_N, M / TILE_M);  // (8, 64)
    gemm_bf16_wmma<true, false, false, false><<<gg, 256, 0, stream>>>(
        xnb, wqb, nullptr, nullptr, nullptr, qb, M, D, D);
    gemm_bf16_wmma<true, false, false, false><<<gg, 256, 0, stream>>>(
        xnb, wkb, nullptr, nullptr, nullptr, kb, M, D, D);
    gemm_bf16_wmma<true, false, false, false><<<gg, 256, 0, stream>>>(
        xnb, wvb, nullptr, nullptr, nullptr, vb, M, D, D);

    // flash attention
    dim3 ga(S / 64, B * H);  // (32, 64)
    flash_attn_kernel<<<ga, 128, 0, stream>>>(qb, kb, vb, att, H, S, D);

    // xt = LN2(xn + att)
    ln_kernel<true><<<M, 256, 0, stream>>>(xn, att, ln2s, ln2b, xt, xtb);

    // h = relu(xt @ W1^T + b1), bf16 out
    gemm_bf16_wmma<true, true, true, false><<<gg, 256, 0, stream>>>(
        xtb, w1b, b1, nullptr, nullptr, hb, M, D, D);

    // out = xt + h @ W2^T + b2, fp32 out
    gemm_bf16_wmma<false, true, false, true><<<gg, 256, 0, stream>>>(
        hb, w2b, b2, xt, (float*)d_out, nullptr, M, D, D);
}
